// KronAE_64836826301092
// MI455X (gfx1250) — matrix operand order repaired and checked by reading, not compiled, not run on hardware
//
#include <hip/hip_runtime.h>
#include <hip/hip_bf16.h>

typedef __attribute__((ext_vector_type(16))) _Float16 v16h;
typedef __attribute__((ext_vector_type(8)))  float    v8f;

#define FTS 128

// ---------------- degree / normalization / layer-0 scalar aggregation --------

__global__ void k_deg(const int* __restrict__ dst, const float* __restrict__ w,
                      float* __restrict__ deg, int E) {
    int e = blockIdx.x * blockDim.x + threadIdx.x;
    if (e < E) atomicAdd(&deg[dst[e]], w[e]);
}

__global__ void k_dinv(float* __restrict__ deg, int N) {
    int n = blockIdx.x * blockDim.x + threadIdx.x;
    if (n < N) {
        float v = deg[n];
        deg[n] = v > 0.f ? rsqrtf(v) : 0.f;   // deg buffer becomes dinv in-place
    }
}

// norm[e] = dinv[src]*w*dinv[dst]; fused layer-0 scalar aggregation:
// agg0[dst] += norm * x[src]   (h0 = x @ W0 is rank-1, so aggregation is scalar)
__global__ void k_norm_agg0(const int* __restrict__ src, const int* __restrict__ dst,
                            const float* __restrict__ w, const float* __restrict__ dinv,
                            const float* __restrict__ x,
                            float* __restrict__ normv, float* __restrict__ agg0, int E) {
    int e = blockIdx.x * blockDim.x + threadIdx.x;
    if (e < E) {
        int s = src[e], d = dst[e];
        float nm = dinv[s] * w[e] * dinv[d];
        normv[e] = nm;
        atomicAdd(&agg0[d], nm * x[s]);
    }
}

__global__ void k_map(const int* __restrict__ idxs, int* __restrict__ map, int Nt) {
    int i = blockIdx.x * blockDim.x + threadIdx.x;
    if (i < Nt) map[idxs[i]] = i;
}

// ------------- pack a 128x128 weight into WMMA 16-bit B-matrix layout --------
// pack[((tn*4+kk)*32 + lane)*16 + e] = B[kk*32 + ((lane&16)?8:0) + (e&7) + ((e&8)?16:0)][tn*16 + (lane&15)]
// transA=1: B[k][n] = W[n*128+k] (i.e. B = Wᵀ, used for X @ FCWᵀ)
// transA=0: B[k][n] = W[k*128+n]
__global__ void k_pack(const float* __restrict__ W, _Float16* __restrict__ dstp, int transA) {
    int t = blockIdx.x * blockDim.x + threadIdx.x;   // 0 .. 16383
    int e  = t & 15;
    int L  = (t >> 4) & 31;
    int kk = (t >> 9) & 3;
    int tn = t >> 11;
    int K = kk * 32 + ((L & 16) ? 8 : 0) + (e & 7) + ((e & 8) ? 16 : 0);
    int n = tn * 16 + (L & 15);
    float v = transA ? W[n * FTS + K] : W[K * FTS + n];
    dstp[t] = (_Float16)v;
}

// ----------------------- WMMA GEMM: out[M,128] = A[M,128] @ B -----------------
// Packed B (32KB) is staged in LDS once per block; each wave computes a 16x128
// output strip: A tile loaded/converted once per K-step, reused across 8 column
// tiles (8 accumulators, 32 WMMA issues per wave, B fed from ds_load).
// fused=1: A[i][k] = relu(agg0[idxs[i]]*W0[k] + b0[k]) synthesized on the fly.
__global__ void k_gemm(const float* __restrict__ A,
                       const float* __restrict__ agg0, const int* __restrict__ idxs,
                       const float* __restrict__ W0, const float* __restrict__ b0,
                       const _Float16* __restrict__ packB, const float* __restrict__ bias,
                       float* __restrict__ out, int M, int do_relu, int fused) {
    __shared__ __align__(32) _Float16 sB[16384];          // whole packed B: 32 KB

    // cooperative stage of packed B into LDS (2048 x 16B)
    {
        const uint4* __restrict__ g = (const uint4*)packB;
        uint4* l = (uint4*)sB;
        for (int i = threadIdx.x; i < 2048; i += blockDim.x) l[i] = g[i];
    }
    __syncthreads();

    int wave = (blockIdx.x * blockDim.x + threadIdx.x) >> 5;
    int lane = threadIdx.x & 31;
    int m0 = wave * 16;
    if (m0 >= M) return;                      // wave-uniform; after the barrier

    int arow = m0 + (lane & 15);
    if (arow >= M) arow = M - 1;              // clamp (dup rows; stores guarded)

    float s = 0.f;
    const float* Abase = A;
    if (fused) s = agg0[idxs[arow]];
    else       Abase = A + (size_t)arow * FTS;

    v8f c[8] = {};
#pragma unroll
    for (int kk = 0; kk < 4; ++kk) {
        // 16-bit A layout: lane (m = lane&15) holds two contiguous 8-deep K
        // runs [kb, kb+8) and [kb+16, kb+24), kb = kk*32 + ((lane&16)?8:0)
        int kb = kk * 32 + ((lane & 16) ? 8 : 0);
        v16h a;
        if (fused) {
#pragma unroll
            for (int i = 0; i < 8; ++i) {
                float v0 = s * W0[kb + i]      + b0[kb + i];
                float v1 = s * W0[kb + 16 + i] + b0[kb + 16 + i];
                a[i]     = (_Float16)(v0 > 0.f ? v0 : 0.f);
                a[8 + i] = (_Float16)(v1 > 0.f ? v1 : 0.f);
            }
        } else {
            const float* ap = Abase + kb;
#pragma unroll
            for (int i = 0; i < 8; ++i) {
                a[i]     = (_Float16)ap[i];
                a[8 + i] = (_Float16)ap[16 + i];
            }
        }
#pragma unroll
        for (int tn = 0; tn < 8; ++tn) {
            v16h b = *(const v16h*)(sB + (((tn * 4 + kk) * 32 + lane) << 4));
            c[tn] = __builtin_amdgcn_wmma_f32_16x16x32_f16(false, a, false, b,
                                                           (short)0, c[tn], false, false);
        }
    }

    int roff = (lane & 16) ? 8 : 0;
#pragma unroll
    for (int tn = 0; tn < 8; ++tn) {
        int col = tn * 16 + (lane & 15);
        float bcol = bias ? bias[col] : 0.f;
#pragma unroll
        for (int r = 0; r < 8; ++r) {
            int row = m0 + roff + r;
            if (row < M) {
                float v = c[tn][r] + bcol;
                if (do_relu) v = v > 0.f ? v : 0.f;
                out[(size_t)row * FTS + col] = v;
            }
        }
    }
}

// -------- layer-1 edge aggregation: Acc[dst,:] += norm * Hc[map[src],:] ------
// One wave per edge; each lane handles 4 features (float4 load + 4 f32 atomics).
__global__ void k_agg1(const int* __restrict__ src, const int* __restrict__ dst,
                       const float* __restrict__ normv, const int* __restrict__ map,
                       const float* __restrict__ Hc, float* __restrict__ Acc, int E) {
    int e = (blockIdx.x * blockDim.x + threadIdx.x) >> 5;
    int lane = threadIdx.x & 31;
    if (e >= E) return;
    int mi = map[src[e]];
    if (mi < 0) return;                 // source row of X3 is all-zero
    float nm = normv[e];
    const float4 hv = *(const float4*)(Hc + (size_t)mi * FTS + lane * 4);
    float* ab = Acc + (size_t)dst[e] * FTS + lane * 4;
    atomicAdd(ab + 0, nm * hv.x);
    atomicAdd(ab + 1, nm * hv.y);
    atomicAdd(ab + 2, nm * hv.z);
    atomicAdd(ab + 3, nm * hv.w);
}

// h2[n] = sum_f relu(Acc[n,f] + b1[f]) * W2[f]   (fused X4 + GEMV), wave/node
__global__ void k_x4h2(const float* __restrict__ Acc, const float* __restrict__ b1,
                       const float* __restrict__ W2, float* __restrict__ h2, int N) {
    int n = (blockIdx.x * blockDim.x + threadIdx.x) >> 5;
    int lane = threadIdx.x & 31;
    if (n >= N) return;
    const float4 av = *(const float4*)(Acc + (size_t)n * FTS + lane * 4);
    const float4 bv = *(const float4*)(b1 + lane * 4);
    const float4 wv = *(const float4*)(W2 + lane * 4);
    float p = 0.f, t;
    t = av.x + bv.x; p += (t > 0.f ? t : 0.f) * wv.x;
    t = av.y + bv.y; p += (t > 0.f ? t : 0.f) * wv.y;
    t = av.z + bv.z; p += (t > 0.f ? t : 0.f) * wv.z;
    t = av.w + bv.w; p += (t > 0.f ? t : 0.f) * wv.w;
#pragma unroll
    for (int off = 16; off > 0; off >>= 1) p += __shfl_xor(p, off, 32);
    if (lane == 0) h2[n] = p;
}

__global__ void k_out_edges(const int* __restrict__ src, const int* __restrict__ dst,
                            const float* __restrict__ normv, const float* __restrict__ h2,
                            float* __restrict__ oacc, int E) {
    int e = blockIdx.x * blockDim.x + threadIdx.x;
    if (e < E) atomicAdd(&oacc[dst[e]], normv[e] * h2[src[e]]);
}

__global__ void k_finalize(const float* __restrict__ oacc, const float* __restrict__ b2,
                           float* __restrict__ out, int N) {
    int n = blockIdx.x * blockDim.x + threadIdx.x;
    if (n < N) out[n] = oacc[n] + b2[0];
}

// -----------------------------------------------------------------------------

extern "C" void kernel_launch(void* const* d_in, const int* in_sizes, int n_in,
                              void* d_out, int out_size, void* d_ws, size_t ws_size,
                              hipStream_t stream) {
    const float* x      = (const float*)d_in[0];
    const int*   edges  = (const int*)  d_in[1];
    const float* ew     = (const float*)d_in[2];
    const int*   idxs_t = (const int*)  d_in[3];
    const float* W0     = (const float*)d_in[4];
    const float* b0     = (const float*)d_in[5];
    const float* W1     = (const float*)d_in[6];
    const float* b1     = (const float*)d_in[7];
    const float* W2     = (const float*)d_in[8];
    const float* b2     = (const float*)d_in[9];
    const float* FCW    = (const float*)d_in[10];
    const float* FCb    = (const float*)d_in[11];

    const int N  = in_sizes[0];
    const int E  = in_sizes[2];
    const int Nt = in_sizes[3];
    const int* src = edges;
    const int* dst = edges + E;
    (void)n_in; (void)out_size; (void)ws_size;

    char* ws = (char*)d_ws;
    size_t off = 0;
    auto take = [&](size_t bytes) -> char* {
        char* p = ws + off;
        off = (off + bytes + 511) & ~(size_t)511;
        return p;
    };
    float*    deg   = (float*)take((size_t)N * 4);        // becomes dinv in-place
    float*    agg0  = (float*)take((size_t)N * 4);
    float*    normv = (float*)take((size_t)E * 4);
    int*      map   = (int*)  take((size_t)N * 4);
    float*    h2    = (float*)take((size_t)N * 4);
    float*    oacc  = (float*)take((size_t)N * 4);
    float*    X2    = (float*)take((size_t)Nt * FTS * 4);
    float*    Hc    = (float*)take((size_t)Nt * FTS * 4);
    float*    Acc   = (float*)take((size_t)N * FTS * 4);
    _Float16* pFC   = (_Float16*)take(16384 * 2);
    _Float16* pW1   = (_Float16*)take(16384 * 2);

    hipMemsetAsync(deg,  0,    (size_t)N * 4, stream);
    hipMemsetAsync(agg0, 0,    (size_t)N * 4, stream);
    hipMemsetAsync(oacc, 0,    (size_t)N * 4, stream);
    hipMemsetAsync(Acc,  0,    (size_t)N * FTS * 4, stream);
    hipMemsetAsync(map,  0xFF, (size_t)N * 4, stream);   // -1

    const int B = 256;
    k_deg      <<<(E + B - 1) / B, B, 0, stream>>>(dst, ew, deg, E);
    k_dinv     <<<(N + B - 1) / B, B, 0, stream>>>(deg, N);
    k_norm_agg0<<<(E + B - 1) / B, B, 0, stream>>>(src, dst, ew, deg, x, normv, agg0, E);
    k_map      <<<(Nt + B - 1) / B, B, 0, stream>>>(idxs_t, map, Nt);
    k_pack     <<<16384 / B, B, 0, stream>>>(FCW, pFC, 1);   // B = FCWᵀ
    k_pack     <<<16384 / B, B, 0, stream>>>(W1,  pW1, 0);   // B = W1

    {
        int waves = (Nt + 15) / 16;                // one wave per 16x128 strip
        int thr   = waves * 32;
        int blk   = (thr + B - 1) / B;
        // X2 = relu( relu(agg0*W0+b0)[rows] @ FCWᵀ + FCb ), A synthesized (fused)
        k_gemm<<<blk, B, 0, stream>>>(nullptr, agg0, idxs_t, W0, b0,
                                      pFC, FCb, X2, Nt, 1, 1);
        // Hc = X2 @ W1
        k_gemm<<<blk, B, 0, stream>>>(X2, nullptr, nullptr, nullptr, nullptr,
                                      pW1, nullptr, Hc, Nt, 0, 0);
    }
    {
        long long thr = (long long)E * 32;
        k_agg1<<<(unsigned)((thr + B - 1) / B), B, 0, stream>>>(src, dst, normv, map, Hc, Acc, E);
    }
    {
        long long thr = (long long)N * 32;
        k_x4h2<<<(unsigned)((thr + B - 1) / B), B, 0, stream>>>(Acc, b1, W2, h2, N);
    }
    k_out_edges<<<(E + B - 1) / B, B, 0, stream>>>(src, dst, normv, h2, oacc, E);
    k_finalize <<<(N + B - 1) / B, B, 0, stream>>>(oacc, b2, (float*)d_out, N);
}
